// IntPointwiseConv1D_3521873183455
// MI455X (gfx1250) — 8-bit matrix operand order repaired and checked by reading, compiled, not run on hardware
//
#include <hip/hip_runtime.h>

typedef int v8i __attribute__((ext_vector_type(8)));

#define BATCH   8
#define IN_CH   256
#define OUT_CH  256
#define LEN     4096
#define KCHUNKS 4
#define NTILES  (LEN / 16)
#define MTILES  (OUT_CH / 16)

__device__ __forceinline__ int clamp_i8(int v) {
    v = v < -128 ? -128 : v;
    return v > 127 ? 127 : v;
}

__global__ __launch_bounds__(256) void pack_w_kernel(const int* __restrict__ w,
                                                     int* __restrict__ wp) {
    const int lane = threadIdx.x & 31;
    const int frag = blockIdx.x * 8 + (threadIdx.x >> 5);
    const int mt = frag >> 2;
    const int kc = frag & 3;
    const int half = lane >> 4;
    const int m    = lane & 15;
    const int o    = mt * 16 + m;
#pragma unroll
    for (int j = 0; j < 8; ++j) {
        const int k0 = kc * 64 + (j >> 2) * 32 + half * 16 + (j & 3) * 4;
        unsigned d = 0;
#pragma unroll
        for (int bb = 0; bb < 4; ++bb) {
            const int v = w[o * IN_CH + k0 + bb];
            d |= (unsigned)(v & 0xFF) << (8 * bb);
        }
        wp[frag * 256 + lane * 8 + j] = (int)d;
    }
}

__global__ __launch_bounds__(256) void pack_x_kernel(const int* __restrict__ x,
                                                     int* __restrict__ xp) {
    const int lane = threadIdx.x & 31;
    const int frag = blockIdx.x * 8 + (threadIdx.x >> 5);
    const int kc = frag & 3;
    const int nt = (frag >> 2) & (NTILES - 1);
    const int b  = frag >> 10;
    const int half = lane >> 4;
    const int n    = lane & 15;
    const int l    = nt * 16 + n;
#pragma unroll
    for (int j = 0; j < 8; ++j) {
        const int k0 = kc * 64 + (j >> 2) * 32 + half * 16 + (j & 3) * 4;
        unsigned d = 0;
#pragma unroll
        for (int bb = 0; bb < 4; ++bb) {
            const int v = x[(b * IN_CH + k0 + bb) * LEN + l];
            d |= (unsigned)(v & 0xFF) << (8 * bb);
        }
        xp[frag * 256 + lane * 8 + j] = (int)d;
    }
}

__global__ __launch_bounds__(256) void gemm_kernel(const int* __restrict__ wp,
                                                   const int* __restrict__ xp,
                                                   const int* __restrict__ shift_p,
                                                   float* __restrict__ out) {
    const int shift = shift_p[0];
    const int th    = 1 << shift;

    const int lane = threadIdx.x & 31;
    const int g    = blockIdx.x * 8 + (threadIdx.x >> 5);
    const int ntg  = g & 63;
    const int mt   = (g >> 6) & 15;
    const int b    = g >> 10;
    const int half = lane >> 4;
    const int ncol = lane & 15;

    v8i a[KCHUNKS];
#pragma unroll
    for (int kc = 0; kc < KCHUNKS; ++kc) {
        a[kc] = *(const v8i*)(wp + (mt * 4 + kc) * 256 + lane * 8);
    }

#pragma unroll
    for (int ni = 0; ni < 4; ++ni) {
        const int nt = ntg * 4 + ni;
        v8i c = {};
#pragma unroll
        for (int kc = 0; kc < KCHUNKS; ++kc) {
            const v8i bfrag =
                *(const v8i*)(xp + ((b * NTILES + nt) * 4 + kc) * 256 + lane * 8);
            c = __builtin_amdgcn_wmma_i32_16x16x64_iu8(
                     true, a[kc],  true, bfrag, c,
                     false,  false);
        }
        const int lcol = nt * 16 + ncol;
#pragma unroll
        for (int r = 0; r < 8; ++r) {
            int acc  = c[r];
            int keep = (acc >= th) || (acc <= -th);
            int v    = keep ? (acc >> shift) : 0;
            v        = clamp_i8(v);
            const int m = r + half * 8;
            out[(size_t)((b * OUT_CH + mt * 16 + m)) * LEN + lcol] = (float)v;
        }
    }
}

extern "C" void kernel_launch(void* const* d_in, const int* in_sizes, int n_in,
                              void* d_out, int out_size, void* d_ws, size_t ws_size,
                              hipStream_t stream) {
    const int* x      = (const int*)d_in[0];
    const int* w      = (const int*)d_in[1];
    const int* shift  = (const int*)d_in[2];
    float*     out    = (float*)d_out;

    int* wp = (int*)d_ws;
    int* xp = (int*)((char*)d_ws + 65536);

    pack_w_kernel<<<8, 256, 0, stream>>>(w, wp);
    pack_x_kernel<<<1024, 256, 0, stream>>>(x, xp);
    gemm_kernel<<<1024, 256, 0, stream>>>(wp, xp, shift, out);
}
